// INCWeightOnlyLinear_86071144611916
// MI455X (gfx1250) — matrix operand order repaired and checked by reading, not compiled, not run on hardware
//
#include <hip/hip_runtime.h>

typedef __attribute__((ext_vector_type(16))) _Float16 v16h;
typedef __attribute__((ext_vector_type(8)))  _Float16 v8h;
typedef __attribute__((ext_vector_type(2)))  _Float16 v2h;
typedef __attribute__((ext_vector_type(8)))  float    v8f;
typedef __attribute__((ext_vector_type(4)))  unsigned v4u;
typedef __attribute__((ext_vector_type(8)))  int      v8i;
typedef __attribute__((ext_vector_type(4)))  int      v4i;

#define IN_F   4096
#define OUT_F  4096
#define BM     128
#define BN     128
#define BK     32
#define LDA    40      // padded LDS stride (halfs) -> conflict-free frag reads
#define LDB    40
#define GROUP  128
#define NK     (IN_F / BK)

#define HAS_TDM __has_builtin(__builtin_amdgcn_tensor_load_to_lds)

union V16 { v16h v; v8h h[2]; };
union H2U { unsigned u; v2h h; };

// int4 pair -> f16 pair via v_perm + mantissa trick:
// 0x6400 | n == f16(1024+n) exactly; (1024+n)-(1024+zp) is exact in f16.
__device__ __forceinline__ v8h dequant8(unsigned q, v2h zoff, v2h sch) {
    const unsigned lo = q & 0x0F0F0F0Fu;
    const unsigned hi = (q >> 4) & 0x0F0F0F0Fu;
    H2U p0, p1, p2, p3;
    p0.u = __builtin_amdgcn_perm(hi, lo, 0x0C040C00u) | 0x64006400u; // {n0,n1}
    p1.u = __builtin_amdgcn_perm(hi, lo, 0x0C050C01u) | 0x64006400u; // {n2,n3}
    p2.u = __builtin_amdgcn_perm(hi, lo, 0x0C060C02u) | 0x64006400u; // {n4,n5}
    p3.u = __builtin_amdgcn_perm(hi, lo, 0x0C070C03u) | 0x64006400u; // {n6,n7}
    const v2h w0 = (p0.h - zoff) * sch;
    const v2h w1 = (p1.h - zoff) * sch;
    const v2h w2 = (p2.h - zoff) * sch;
    const v2h w3 = (p3.h - zoff) * sch;
    v8h r;
    r[0]=w0[0]; r[1]=w0[1]; r[2]=w1[0]; r[3]=w1[1];
    r[4]=w2[0]; r[5]=w2[1]; r[6]=w3[0]; r[7]=w3[1];
    return r;
}

#if HAS_TDM
// One-shot TDM DMA: contiguous 8KB B-tile (4096 f16) global -> LDS, with
// 16B LDS padding after every 64B (pad_interval=16 DW, pad_amount=4 DW)
// reproducing the stride-40-half bank-conflict-free layout.
__device__ __forceinline__ void tdm_load_b_tile(const _Float16* gsrc, unsigned lds_off) {
    const unsigned long long ga = (unsigned long long)(uintptr_t)gsrc;
    v4u g0;
    g0[0] = 1u;                                            // count=1, no gather
    g0[1] = lds_off;                                       // lds_addr (bytes)
    g0[2] = (unsigned)(ga & 0xFFFFFFFFu);                  // global_addr[31:0]
    g0[3] = (unsigned)((ga >> 32) & 0x01FFFFFFu)           // global_addr[56:32]
          | 0x80000000u;                                   // type=2 (image)
    v8i g1;
    g1[0] = 0x06D10000;  // pad_amount=4DW<<25 | pad_int=16DW<<22 | pad_en | data_size=2B
    g1[1] = 0x10000000;  // tensor_dim0 = 4096 (bits[63:48])
    g1[2] = 0x00010000;  // tensor_dim1 = 1    (bits[95:80])
    g1[3] = 0x10000000;  // tile_dim0  = 4096  (bits[127:112])
    g1[4] = 0x00000001;  // tile_dim1  = 1
    g1[5] = 0x00001000;  // tensor_dim0_stride = 4096
    g1[6] = 0;
    g1[7] = 0;
    const v4i z4 = {0, 0, 0, 0};
    const v8i z8 = {0, 0, 0, 0, 0, 0, 0, 0};
    // 6-arg toolchain form: (g0, g1, g2, g3, ext, cpol)
    __builtin_amdgcn_tensor_load_to_lds(g0, g1, z4, z4, z8, 0);
}
#endif

// ---------------- Phase 1: dequantize W once into tiled f16 ----------------
// wt layout: [N/128][K/32][nc=128][kc=32] f16 -> each GEMM B-tile is one
// contiguous 8KB block. Threads t..t+3 emit 64B contiguous stores.
__global__ __launch_bounds__(256) void wq4_dequant_w(
    const int*   __restrict__ qweight,   // [512, 4096]
    const int*   __restrict__ qzeros,    // [32, 512]
    const float* __restrict__ scales,    // [32, 4096]
    _Float16*    __restrict__ wt)
{
    const int t = threadIdx.x;
    const int n = blockIdx.x * 64 + (t >> 2);
    const int r = blockIdx.y * 4 + (t & 3);
    const int g = r >> 4;

    const float    sc = scales[g * OUT_F + n];
    const unsigned zq = ((unsigned)qzeros[g * (OUT_F / 8) + (n >> 3)]
                         >> ((n & 7) * 4)) & 0xFu;
    const _Float16 zh = (_Float16)(float)(1025u + zq);   // 1024 + (zq+1)
    const _Float16 sh = (_Float16)sc;
    const v2h zoff = {zh, zh};
    const v2h sch  = {sh, sh};

    const unsigned q = (unsigned)qweight[(size_t)r * OUT_F + n];
    const v8h w = dequant8(q, zoff, sch);

    const int k = r << 3;
    _Float16* dst = wt + ((size_t)(n >> 7) * (IN_F / BK) + (k >> 5)) * (BN * BK)
                       + (n & 127) * BK + (k & 31);
    *(v8h*)dst = w;
}

// ---------------- Phase 2: pure f16 WMMA GEMM, double-buffered LDS ----------
// B tiles staged by the Tensor Data Mover (wave 0 issues, TENSORcnt-synced);
// A tiles staged by all threads with fp32->f16 conversion.
__global__ __launch_bounds__(256) void wq4_gemm_phase2(
    const float*    __restrict__ x,
    const _Float16* __restrict__ wt,
    const float*    __restrict__ bias,
    float*          __restrict__ out)
{
    __shared__ _Float16 Alds[2][BM * LDA];
    __shared__ _Float16 Blds[2][BN * LDB];

    const int t    = threadIdx.x;
    const int lane = t & 31;
    const int wave = t >> 5;
    const int wr   = wave >> 2;
    const int wc   = wave & 3;
    const int lrow = lane & 15;
    const int lhi  = lane >> 4;

    const int m0 = blockIdx.y * BM;
    const int n0 = blockIdx.x * BN;

    const int arow = t >> 1;
    const int acol = (t & 1) << 4;
    const float* xrow = x + (size_t)(m0 + arow) * IN_F + acol;

    const _Float16* wblk = wt + (size_t)blockIdx.x * (IN_F / BK) * (BN * BK);

    float4 fa0, fa1, fa2, fa3;
#if !HAS_TDM
    const int bnc = t >> 1;
    const int bkc = (t & 1) << 4;
    uint4 qb0, qb1;
#endif

#define FETCH_A(kt) do { \
    const float4* xp_ = reinterpret_cast<const float4*>(xrow + (kt) * BK); \
    fa0 = xp_[0]; fa1 = xp_[1]; fa2 = xp_[2]; fa3 = xp_[3]; \
} while (0)

#define STAGE_A(buf) do { \
    v8h ha_, hb_; \
    ha_[0]=(_Float16)fa0.x; ha_[1]=(_Float16)fa0.y; ha_[2]=(_Float16)fa0.z; ha_[3]=(_Float16)fa0.w; \
    ha_[4]=(_Float16)fa1.x; ha_[5]=(_Float16)fa1.y; ha_[6]=(_Float16)fa1.z; ha_[7]=(_Float16)fa1.w; \
    hb_[0]=(_Float16)fa2.x; hb_[1]=(_Float16)fa2.y; hb_[2]=(_Float16)fa2.z; hb_[3]=(_Float16)fa2.w; \
    hb_[4]=(_Float16)fa3.x; hb_[5]=(_Float16)fa3.y; hb_[6]=(_Float16)fa3.z; hb_[7]=(_Float16)fa3.w; \
    *(v8h*)&Alds[buf][arow * LDA + acol]     = ha_; \
    *(v8h*)&Alds[buf][arow * LDA + acol + 8] = hb_; \
} while (0)

#if !HAS_TDM
#define FETCH_B(kt) do { \
    const uint4* bp_ = reinterpret_cast<const uint4*>(wblk + (size_t)(kt) * (BN * BK)) + (t << 1); \
    qb0 = bp_[0]; qb1 = bp_[1]; \
} while (0)
#define STAGE_B(buf) do { \
    *(uint4*)&Blds[buf][bnc * LDB + bkc]     = qb0; \
    *(uint4*)&Blds[buf][bnc * LDB + bkc + 8] = qb1; \
} while (0)
#endif

    v8f acc[4][2] = {};

    // ---- prologue: tile 0 ----
    FETCH_A(0);
#if HAS_TDM
    if (wave == 0)
        tdm_load_b_tile(wblk, (unsigned)(uintptr_t)&Blds[0][0]);
#else
    FETCH_B(0);
#endif
    STAGE_A(0);
#if HAS_TDM
    if (wave == 0) __builtin_amdgcn_s_wait_tensorcnt(0);
#else
    STAGE_B(0);
#endif
    __syncthreads();

    for (int kt = 0; kt < NK; ++kt) {
        const int cur = kt & 1;
        if (kt + 1 < NK) {
            FETCH_A(kt + 1);
#if HAS_TDM
            if (wave == 0)
                tdm_load_b_tile(wblk + (size_t)(kt + 1) * (BN * BK),
                                (unsigned)(uintptr_t)&Blds[cur ^ 1][0]);
#else
            FETCH_B(kt + 1);
#endif
        }

        v16h afrag[4];
        #pragma unroll
        for (int mt = 0; mt < 4; ++mt) {
            const int row = wr * 64 + mt * 16 + lrow;
            V16 u;
            u.h[0] = *(const v8h*)&Alds[cur][row * LDA + lhi * 16];
            u.h[1] = *(const v8h*)&Alds[cur][row * LDA + lhi * 16 + 8];
            afrag[mt] = u.v;
        }
        v16h bfrag[2];
        #pragma unroll
        for (int nt = 0; nt < 2; ++nt) {
            const int col = wc * 32 + nt * 16 + lrow;
            V16 u;
            u.h[0] = *(const v8h*)&Blds[cur][col * LDB + lhi * 16];
            u.h[1] = *(const v8h*)&Blds[cur][col * LDB + lhi * 16 + 8];
            bfrag[nt] = u.v;
        }
        #pragma unroll
        for (int mt = 0; mt < 4; ++mt)
            #pragma unroll
            for (int nt = 0; nt < 2; ++nt)
                acc[mt][nt] = __builtin_amdgcn_wmma_f32_16x16x32_f16(
                    false, afrag[mt], false, bfrag[nt],
                    (short)0, acc[mt][nt], false, false);

        if (kt + 1 < NK) STAGE_A(cur ^ 1);
#if HAS_TDM
        if (wave == 0) __builtin_amdgcn_s_wait_tensorcnt(0);
#else
        if (kt + 1 < NK) STAGE_B(cur ^ 1);
#endif
        __syncthreads();
    }
#undef FETCH_A
#undef STAGE_A
#if !HAS_TDM
#undef FETCH_B
#undef STAGE_B
#endif

    #pragma unroll
    for (int mt = 0; mt < 4; ++mt) {
        #pragma unroll
        for (int nt = 0; nt < 2; ++nt) {
            const int   col   = n0 + wc * 32 + nt * 16 + lrow;
            const float bv    = bias[col];
            const int   rbase = m0 + wr * 64 + mt * 16 + lhi * 8;
            #pragma unroll
            for (int j = 0; j < 8; ++j)
                out[(size_t)(rbase + j) * OUT_F + col] = acc[mt][nt][j] + bv;
        }
    }
}

// ---------------- Fallback: fused dequant + GEMM (small ws) -----------------
__global__ __launch_bounds__(256) void wq4_gemm_fused(
    const float* __restrict__ x,
    const int*   __restrict__ qweight,
    const int*   __restrict__ qzeros,
    const float* __restrict__ scales,
    const float* __restrict__ bias,
    float* __restrict__ out)
{
    __shared__ _Float16 Alds[BM * LDA];
    __shared__ _Float16 Blds[BN * LDB];

    const int t    = threadIdx.x;
    const int lane = t & 31;
    const int wave = t >> 5;
    const int wr   = wave >> 2;
    const int wc   = wave & 3;
    const int lrow = lane & 15;
    const int lhi  = lane >> 4;

    const int m0 = blockIdx.y * BM;
    const int n0 = blockIdx.x * BN;

    const int arow = t >> 1;
    const int acol = (t & 1) << 4;
    const float* xrow = x + (size_t)(m0 + arow) * IN_F;

    const int bn   = t & 127;
    const int brp  = t >> 7;
    const int gcol = n0 + bn;

    v8f acc[4][2] = {};

    for (int g = 0; g < IN_F / GROUP; ++g) {
        const float    sc = scales[g * OUT_F + gcol];
        const unsigned zq = ((unsigned)qzeros[g * (OUT_F / 8) + (gcol >> 3)]
                             >> ((gcol & 7) * 4)) & 0xFu;
        const _Float16 zh = (_Float16)(float)(1025u + zq);
        const _Float16 sh = (_Float16)sc;
        const v2h zoff = {zh, zh};
        const v2h sch  = {sh, sh};

        #pragma unroll
        for (int kk = 0; kk < GROUP / BK; ++kk) {
            const int k0 = g * GROUP + kk * BK;

            const float4* xp = reinterpret_cast<const float4*>(xrow + k0 + acol);
            const float4 f0 = xp[0], f1 = xp[1], f2 = xp[2], f3 = xp[3];
            __builtin_prefetch(xrow + k0 + 2 * BK, 0, 1);

            v8h ha, hb;
            ha[0]=(_Float16)f0.x; ha[1]=(_Float16)f0.y; ha[2]=(_Float16)f0.z; ha[3]=(_Float16)f0.w;
            ha[4]=(_Float16)f1.x; ha[5]=(_Float16)f1.y; ha[6]=(_Float16)f1.z; ha[7]=(_Float16)f1.w;
            hb[0]=(_Float16)f2.x; hb[1]=(_Float16)f2.y; hb[2]=(_Float16)f2.z; hb[3]=(_Float16)f2.w;
            hb[4]=(_Float16)f3.x; hb[5]=(_Float16)f3.y; hb[6]=(_Float16)f3.z; hb[7]=(_Float16)f3.w;

            const int krow = (k0 >> 3) + brp * 2;
            const unsigned q0 = (unsigned)qweight[(size_t)krow * OUT_F + gcol];
            const unsigned q1 = (unsigned)qweight[(size_t)(krow + 1) * OUT_F + gcol];
            const v8h w0 = dequant8(q0, zoff, sch);
            const v8h w1 = dequant8(q1, zoff, sch);

            __syncthreads();
            *(v8h*)&Alds[arow * LDA + acol]         = ha;
            *(v8h*)&Alds[arow * LDA + acol + 8]     = hb;
            *(v8h*)&Blds[bn   * LDB + brp * 16]     = w0;
            *(v8h*)&Blds[bn   * LDB + brp * 16 + 8] = w1;
            __syncthreads();

            v16h afrag[4];
            #pragma unroll
            for (int mt = 0; mt < 4; ++mt) {
                const int row = wr * 64 + mt * 16 + lrow;
                V16 u;
                u.h[0] = *(const v8h*)&Alds[row * LDA + lhi * 16];
                u.h[1] = *(const v8h*)&Alds[row * LDA + lhi * 16 + 8];
                afrag[mt] = u.v;
            }
            v16h bfrag[2];
            #pragma unroll
            for (int nt = 0; nt < 2; ++nt) {
                const int col = wc * 32 + nt * 16 + lrow;
                V16 u;
                u.h[0] = *(const v8h*)&Blds[col * LDB + lhi * 16];
                u.h[1] = *(const v8h*)&Blds[col * LDB + lhi * 16 + 8];
                bfrag[nt] = u.v;
            }
            #pragma unroll
            for (int mt = 0; mt < 4; ++mt)
                #pragma unroll
                for (int nt = 0; nt < 2; ++nt)
                    acc[mt][nt] = __builtin_amdgcn_wmma_f32_16x16x32_f16(
                        false, afrag[mt], false, bfrag[nt],
                        (short)0, acc[mt][nt], false, false);
        }
    }

    #pragma unroll
    for (int mt = 0; mt < 4; ++mt) {
        #pragma unroll
        for (int nt = 0; nt < 2; ++nt) {
            const int   col   = n0 + wc * 32 + nt * 16 + lrow;
            const float bv    = bias[col];
            const int   rbase = m0 + wr * 64 + mt * 16 + lhi * 8;
            #pragma unroll
            for (int j = 0; j < 8; ++j)
                out[(size_t)(rbase + j) * OUT_F + col] = acc[mt][nt][j] + bv;
        }
    }
}

extern "C" void kernel_launch(void* const* d_in, const int* in_sizes, int n_in,
                              void* d_out, int out_size, void* d_ws, size_t ws_size,
                              hipStream_t stream) {
    const float* x    = (const float*)d_in[0];
    const int*   qw   = (const int*)  d_in[1];
    const int*   qz   = (const int*)  d_in[2];
    const float* sc   = (const float*)d_in[3];
    const float* bias = (const float*)d_in[4];
    float*       out  = (float*)d_out;

    const int tokens = in_sizes[0] / IN_F;
    const dim3 grid(OUT_F / BN, tokens / BM);
    const size_t WT_BYTES = (size_t)IN_F * OUT_F * sizeof(_Float16);  // 32 MB

    if (ws_size >= WT_BYTES) {
        _Float16* wt = (_Float16*)d_ws;
        wq4_dequant_w<<<dim3(OUT_F / 64, (IN_F / 8) / 4), dim3(256), 0, stream>>>(qw, qz, sc, wt);
        wq4_gemm_phase2<<<grid, dim3(256), 0, stream>>>(x, wt, bias, out);
    } else {
        wq4_gemm_fused<<<grid, dim3(256), 0, stream>>>(x, qw, qz, sc, bias, out);
    }
}
